// MultiHeadAttention_23510650978891
// MI455X (gfx1250) — matrix operand order repaired and checked by reading, not compiled, not run on hardware
//
#include <hip/hip_runtime.h>

// ---------------------------------------------------------------------------
// MHA forward, B=2, T=2048, C=1024, H=16, D=64 on gfx1250 (CDNA5, wave32).
// bf16 WMMA everywhere; LDS-tiled GEMMs (double-buffered B panels) and a
// flash-attention kernel with workgroup-shared K/V tiles in LDS.
// ---------------------------------------------------------------------------

typedef __attribute__((ext_vector_type(16))) __bf16 v16bf;
typedef __attribute__((ext_vector_type(8)))  __bf16 v8bf;
typedef __attribute__((ext_vector_type(8)))  float  v8f;

constexpr int Tt  = 2048;   // sequence length
constexpr int Cd  = 1024;   // model dim
constexpr int Hh  = 16;     // heads
constexpr int Dh  = 64;     // head dim
constexpr int BT  = 4096;   // B*T rows
constexpr int C3  = 3072;   // 3*C

__device__ __forceinline__ __bf16 f2bf(float f) {
  unsigned u = __builtin_bit_cast(unsigned, f);
  unsigned short s = (unsigned short)((u + 0x7FFFu + ((u >> 16) & 1u)) >> 16);
  return __builtin_bit_cast(__bf16, s);
}

__device__ __forceinline__ v16bf make_frag(v8bf lo, v8bf hi) {
  union { v16bf v; v8bf h[2]; } u;
  u.h[0] = lo; u.h[1] = hi;
  return u.v;
}

// A-fragment (16xK row-major tile), 16-bit A layout per ISA 7.12.2:
// lane&15 = row M; half 0: K=0..7 & 16..23; half 1: +8.
__device__ __forceinline__ v16bf load_a_frag(const __bf16* base, int stride, int lane) {
  const int row  = lane & 15;
  const int koff = (lane >> 4) * 8;
  const __bf16* p = base + row * stride + koff;
  return make_frag(*(const v8bf*)p, *(const v8bf*)(p + 16));
}

// B-fragment loaded from B^T stored row-major [N][K]:
// lane&15 = column N; half 0: K=0..7 & 16..23; half 1: +8 (matches A).
__device__ __forceinline__ v16bf load_b_frag(const __bf16* baseT, int stride, int lane) {
  const int col  = lane & 15;
  const int koff = (lane >> 4) * 8;
  const __bf16* p = baseT + col * stride + koff;
  return make_frag(*(const v8bf*)p, *(const v8bf*)(p + 16));
}

#define WMMA_BF16(a, b, c) \
  __builtin_amdgcn_wmma_f32_16x16x32_bf16(false, (a), false, (b), (short)0, (c), false, false)

// ------------------------------ helpers ------------------------------------

__global__ void cvt_f32_to_bf16_k(const float* __restrict__ in, __bf16* __restrict__ out, int n) {
  int i = blockIdx.x * blockDim.x + threadIdx.x;
  if (i < n) out[i] = f2bf(in[i]);
}

// in: fp32 [R][Cc] row-major  ->  out: bf16 [Cc][R] row-major (i.e. W^T)
__global__ void transpose_to_bf16_k(const float* __restrict__ in, __bf16* __restrict__ out,
                                    int R, int Cc) {
  int i = blockIdx.x * blockDim.x + threadIdx.x;
  if (i < R * Cc) {
    int r = i / Cc, c = i % Cc;
    out[c * R + r] = f2bf(in[i]);
  }
}

// --------------------- shared GEMM mainloop (128x64/WG) --------------------
// 8 waves/WG. Wave owns a 16x64 tile; the 64x32 B panel is staged to LDS
// (double buffered; padded stride 40 to spread LDS banks).
constexpr int LDSB = 40;

__device__ __forceinline__ void gemm_mainloop(
    const __bf16* __restrict__ A, int lda,      // wave's 16-row A tile base
    const __bf16* __restrict__ Bt, int ldb,     // global B^T at (n0, k=0)
    __bf16* Bs,                                 // LDS [2][64*LDSB]
    int K, int tid, int lane, v8f acc[4]) {
  // stage panel k=0
  {
    const int e0 = tid * 8, col = e0 >> 5, k = e0 & 31;
    *(v8bf*)(Bs + col * LDSB + k) = *(const v8bf*)(Bt + col * ldb + k);
  }
  for (int kb = 0; kb < K; kb += 32) {
    __bf16* cur = Bs + ((kb >> 5) & 1) * (64 * LDSB);
    __syncthreads();                     // panel `cur` ready; prior reads done
    if (kb + 32 < K) {                   // prefetch next panel during WMMAs
      __bf16* nxt = Bs + (((kb >> 5) + 1) & 1) * (64 * LDSB);
      const int e0 = tid * 8, col = e0 >> 5, k = e0 & 31;
      *(v8bf*)(nxt + col * LDSB + k) = *(const v8bf*)(Bt + col * ldb + (kb + 32) + k);
    }
    v16bf a = load_a_frag(A + kb, lda, lane);
#pragma unroll
    for (int j = 0; j < 4; ++j) {
      v16bf b = load_b_frag(cur + j * 16 * LDSB, LDSB, lane);
      acc[j] = WMMA_BF16(a, b, acc[j]);
    }
  }
}

// ------------------------------ QKV GEMM -----------------------------------
// Epilogue scatters into Q/K [B,H,T,D] and Vt [B,H,D,T].
__global__ void gemm_qkv_k(const __bf16* __restrict__ xb, const __bf16* __restrict__ Wat,
                           const float* __restrict__ b_attn,
                           __bf16* __restrict__ Qo, __bf16* __restrict__ Ko,
                           __bf16* __restrict__ Vt) {
  __shared__ __align__(16) __bf16 Bs[2 * 64 * LDSB];
  const int tid  = threadIdx.x;
  const int lane = tid & 31;
  const int wv   = tid >> 5;
  const int rtile = blockIdx.x / (C3 / 64);
  const int strip = blockIdx.x % (C3 / 64);
  const int row0 = rtile * 128 + wv * 16;
  const int n0   = strip * 64;

  v8f acc[4] = {};
  gemm_mainloop(xb + (size_t)row0 * Cd, Cd, Wat + (size_t)n0 * Cd, Cd,
                Bs, Cd, tid, lane, acc);

  const int half = lane >> 4;
#pragma unroll
  for (int j = 0; j < 4; ++j) {
#pragma unroll
    for (int r = 0; r < 8; ++r) {
      int row = row0 + r + 8 * half;       // global (b*T + t)
      int n   = n0 + j * 16 + (lane & 15); // column in [0,3C)
      float v = acc[j][r] + b_attn[n];
      int b = row / Tt, t = row % Tt;
      int sec = n / Cd, w = n % Cd;
      int h = w / Dh, d = w % Dh;
      __bf16 bv = f2bf(v);
      if (sec == 0)      Qo[(((b * Hh + h) * Tt) + t) * Dh + d] = bv;
      else if (sec == 1) Ko[(((b * Hh + h) * Tt) + t) * Dh + d] = bv;
      else               Vt[(((b * Hh + h) * Dh) + d) * Tt + t] = bv;
    }
  }
}

// --------------------------- Flash attention -------------------------------
// 4 waves/WG own 64 consecutive query rows of one (b,h); each 32-key step
// stages K (32x64) and V^T (64x32) tiles into LDS shared by all waves.
constexpr int KSTR = 72;   // padded LDS stride for K tile rows (keys)
constexpr int VSTR = 40;   // padded LDS stride for V^T tile rows (d)
constexpr int PSTR = 40;   // padded LDS stride for P staging rows

__global__ void flash_attn_k(const __bf16* __restrict__ Q, const __bf16* __restrict__ K,
                             const __bf16* __restrict__ Vt, __bf16* __restrict__ O) {
  __shared__ __align__(16) __bf16 Ks[32 * KSTR];
  __shared__ __align__(16) __bf16 Vs[64 * VSTR];
  __shared__ __align__(16) __bf16 Ps[4][16 * PSTR];

  const int tid  = threadIdx.x;
  const int lane = tid & 31;
  const int wv   = tid >> 5;
  const int bh = blockIdx.x >> 5;          // [0, B*H)
  const int q0 = (blockIdx.x & 31) << 6;   // 64-row query block
  const int qm = q0 + wv * 16;             // this wave's 16-row tile

  const __bf16* Qb = Q  + (size_t)(bh * Tt + qm) * Dh;
  const __bf16* Kb = K  + (size_t)bh * Tt * Dh;
  const __bf16* Vb = Vt + (size_t)bh * Dh * Tt;

  const v16bf qa0 = load_a_frag(Qb + 0,  Dh, lane);
  const v16bf qa1 = load_a_frag(Qb + 32, Dh, lane);

  float m[8], l[8];
  v8f o[4] = {};
#pragma unroll
  for (int r = 0; r < 8; ++r) { m[r] = -1e30f; l[r] = 0.0f; }

  const int half = lane >> 4;
  const int kend = q0 + 64;                // block-uniform loop bound

  for (int kt = 0; kt < kend; kt += 32) {
    __syncthreads();                       // previous tiles fully consumed
    {
      // stage K tile: 32 keys x 64 d (2048 elems, 16/thread)
      const int e0 = tid * 16, krow = e0 >> 6, d = e0 & 63;
      const __bf16* src = Kb + (size_t)(kt + krow) * Dh + d;
      *(v8bf*)(Ks + krow * KSTR + d)     = *(const v8bf*)(src);
      *(v8bf*)(Ks + krow * KSTR + d + 8) = *(const v8bf*)(src + 8);
      // stage V^T tile: 64 d x 32 keys (2048 elems, 16/thread)
      const int vd = e0 >> 5, kc = e0 & 31;
      const __bf16* vsrc = Vb + (size_t)vd * Tt + kt + kc;
      *(v8bf*)(Vs + vd * VSTR + kc)     = *(const v8bf*)(vsrc);
      *(v8bf*)(Vs + vd * VSTR + kc + 8) = *(const v8bf*)(vsrc + 8);
    }
    __syncthreads();                       // tiles visible to all waves

    if (kt < qm + 16) {                    // wave-uniform causal frontier
      v8f s0 = {}, s1 = {};
      {
        v16bf k0 = load_b_frag(Ks + 0,  KSTR, lane);
        v16bf k1 = load_b_frag(Ks + 32, KSTR, lane);
        s0 = WMMA_BF16(qa0, k0, s0);
        s0 = WMMA_BF16(qa1, k1, s0);
      }
      {
        v16bf k0 = load_b_frag(Ks + 16 * KSTR + 0,  KSTR, lane);
        v16bf k1 = load_b_frag(Ks + 16 * KSTR + 32, KSTR, lane);
        s1 = WMMA_BF16(qa0, k0, s1);
        s1 = WMMA_BF16(qa1, k1, s1);
      }

      const int col0 = kt + (lane & 15);
      const int col1 = col0 + 16;
#pragma unroll
      for (int r = 0; r < 8; ++r) {
        const int rowg = qm + r + 8 * half;
        float a = s0[r] * 0.125f; if (col0 > rowg) a = -1e30f;
        float b = s1[r] * 0.125f; if (col1 > rowg) b = -1e30f;
        float mx = fmaxf(a, b);
        mx = fmaxf(mx, __shfl_xor(mx, 1, 32));
        mx = fmaxf(mx, __shfl_xor(mx, 2, 32));
        mx = fmaxf(mx, __shfl_xor(mx, 4, 32));
        mx = fmaxf(mx, __shfl_xor(mx, 8, 32));
        float mn = fmaxf(m[r], mx);
        float alpha = __expf(m[r] - mn);
        float p0 = __expf(a - mn);
        float p1 = __expf(b - mn);
        float rs = p0 + p1;
        rs += __shfl_xor(rs, 1, 32);
        rs += __shfl_xor(rs, 2, 32);
        rs += __shfl_xor(rs, 4, 32);
        rs += __shfl_xor(rs, 8, 32);
        l[r] = l[r] * alpha + rs;
        m[r] = mn;
#pragma unroll
        for (int t = 0; t < 4; ++t) o[t][r] *= alpha;
        Ps[wv][(r + 8 * half) * PSTR + (lane & 15)]      = f2bf(p0);
        Ps[wv][(r + 8 * half) * PSTR + 16 + (lane & 15)] = f2bf(p1);
      }

      v16bf pA = load_a_frag(&Ps[wv][0], PSTR, lane);
#pragma unroll
      for (int t = 0; t < 4; ++t) {
        v16bf vB = load_b_frag(Vs + (t * 16) * VSTR, VSTR, lane);
        o[t] = WMMA_BF16(pA, vB, o[t]);
      }
    }
  }

  // epilogue -> O [B,T,C] bf16
  const int b = bh / Hh, h = bh % Hh;
#pragma unroll
  for (int t = 0; t < 4; ++t) {
#pragma unroll
    for (int r = 0; r < 8; ++r) {
      int trow = qm + r + 8 * half;
      int col  = h * Dh + t * 16 + (lane & 15);
      O[(size_t)(b * Tt + trow) * Cd + col] = f2bf(o[t][r] / l[r]);
    }
  }
}

// ------------------------------ Proj GEMM ----------------------------------
__global__ void gemm_proj_k(const __bf16* __restrict__ Ob, const __bf16* __restrict__ Wpt,
                            const float* __restrict__ b_proj, float* __restrict__ out) {
  __shared__ __align__(16) __bf16 Bs[2 * 64 * LDSB];
  const int tid  = threadIdx.x;
  const int lane = tid & 31;
  const int wv   = tid >> 5;
  const int rtile = blockIdx.x / (Cd / 64);
  const int strip = blockIdx.x % (Cd / 64);
  const int row0 = rtile * 128 + wv * 16;
  const int n0   = strip * 64;

  v8f acc[4] = {};
  gemm_mainloop(Ob + (size_t)row0 * Cd, Cd, Wpt + (size_t)n0 * Cd, Cd,
                Bs, Cd, tid, lane, acc);

  const int half = lane >> 4;
#pragma unroll
  for (int j = 0; j < 4; ++j) {
#pragma unroll
    for (int r = 0; r < 8; ++r) {
      int row = row0 + r + 8 * half;
      int n   = n0 + j * 16 + (lane & 15);
      out[(size_t)row * Cd + n] = acc[j][r] + b_proj[n];
    }
  }
}

// ------------------------------ launcher -----------------------------------
extern "C" void kernel_launch(void* const* d_in, const int* in_sizes, int n_in,
                              void* d_out, int out_size, void* d_ws, size_t ws_size,
                              hipStream_t stream) {
  (void)in_sizes; (void)n_in; (void)out_size; (void)ws_size;

  const float* x      = (const float*)d_in[0];   // [B,T,C]
  const float* W_attn = (const float*)d_in[1];   // [C,3C]
  const float* b_attn = (const float*)d_in[2];   // [3C]
  const float* W_proj = (const float*)d_in[3];   // [C,C]
  const float* b_proj = (const float*)d_in[4];   // [C]
  float* out = (float*)d_out;                    // [B,T,C] fp32

  char* ws = (char*)d_ws;
  __bf16* xb  = (__bf16*)(ws);                   //  8 MiB: x bf16
  __bf16* Wat = (__bf16*)(ws + (8u  << 20));     //  6 MiB: W_attn^T bf16
  __bf16* Wpt = (__bf16*)(ws + (14u << 20));     //  2 MiB: W_proj^T bf16
  __bf16* Qd  = (__bf16*)(ws + (16u << 20));     //  8 MiB: Q [B,H,T,D]
  __bf16* Kd  = (__bf16*)(ws + (24u << 20));     //  8 MiB: K [B,H,T,D]
  __bf16* Vtd = (__bf16*)(ws + (32u << 20));     //  8 MiB: V^T [B,H,D,T]
  __bf16* Od  = (__bf16*)(ws + (40u << 20));     //  8 MiB: attn out [B,T,C]

  cvt_f32_to_bf16_k<<<(BT * Cd + 255) / 256, 256, 0, stream>>>(x, xb, BT * Cd);
  transpose_to_bf16_k<<<(Cd * C3 + 255) / 256, 256, 0, stream>>>(W_attn, Wat, Cd, C3);
  transpose_to_bf16_k<<<(Cd * Cd + 255) / 256, 256, 0, stream>>>(W_proj, Wpt, Cd, Cd);

  // 32 row blocks (128 rows) * 48 col strips (64 cols), 8 waves/WG
  gemm_qkv_k<<<1536, 256, 0, stream>>>(xb, Wat, b_attn, Qd, Kd, Vtd);
  // 32 (b,h) * 32 query blocks (64 rows), 4 waves/WG
  flash_attn_k<<<1024, 128, 0, stream>>>(Qd, Kd, Vtd, Od);
  // 32 row blocks * 16 col strips, 8 waves/WG
  gemm_proj_k<<<512, 256, 0, stream>>>(Od, Wpt, b_proj, out);
}
